// CAN_70231305224577
// MI455X (gfx1250) — matrix operand order repaired and checked by reading, not compiled, not run on hardware
//
#include <hip/hip_runtime.h>
#include <hip/hip_bf16.h>
#include <stdint.h>

// ---------------------------------------------------------------------------
// Problem constants (from reference setup_inputs)
// ---------------------------------------------------------------------------
#define BI 64
#define RR 36
#define BC 64
#define WW 40
#define DD 512
#define WP 48     // W padded to 3 WMMA M-tiles
#define RP 48     // R padded to 3 WMMA N-tiles (for S)
#define RK 64     // R padded to multiple of 32 for K-dim of attn@v

typedef __bf16 bf16;
typedef __attribute__((ext_vector_type(8)))  __bf16 v8bf;
typedef __attribute__((ext_vector_type(16))) __bf16 v16bf;
typedef __attribute__((ext_vector_type(8)))  float  v8f;

// ---------------------------------------------------------------------------
// WMMA helpers (v_wmma_f32_16x16x32_bf16, wave32)
// ---------------------------------------------------------------------------
__device__ __forceinline__ v8f wmma_bf16(v16bf a, v16bf b, v8f c) {
    return __builtin_amdgcn_wmma_f32_16x16x32_bf16(
        /*neg_a=*/false, a, /*neg_b=*/false, b,
        /*c_mod=*/(short)0, c, /*reuse_a=*/false, /*reuse_b=*/false);
}

// A fragment: 16x32 bf16, row-major source with leading dim ld (elements).
__device__ __forceinline__ v16bf load_a_frag(const bf16* base, int ld,
                                             int mBase, int kBase, int lane) {
    int half = lane >> 4;
    int row  = mBase + (lane & 15);
    const bf16* p0 = base + (size_t)row * ld + kBase + half * 8;
    union { v16bf v; v8bf h[2]; } u;
    u.h[0] = *(const v8bf*)p0;
    u.h[1] = *(const v8bf*)(p0 + 16);
    return u.v;
}

// B fragment: 32x16 bf16 where B[k][n] = baseNK[n*ld + k] (row-major [N,K]).
__device__ __forceinline__ v16bf load_b_frag(const bf16* baseNK, int ld,
                                             int kBase, int nBase, int lane) {
    int col = nBase + (lane & 15);
    const bf16* p = baseNK + (size_t)col * ld + kBase + (lane >> 4) * 8;
    union { v16bf v; v8bf h[2]; } u;
    u.h[0] = *(const v8bf*)p;
    u.h[1] = *(const v8bf*)(p + 16);
    return u.v;
}

__device__ __forceinline__ float wave_sum(float v) {
    for (int off = 16; off > 0; off >>= 1) v += __shfl_xor(v, off, 32);
    return v;
}
__device__ __forceinline__ float wave_max(float v) {
    for (int off = 16; off > 0; off >>= 1) v = fmaxf(v, __shfl_xor(v, off, 32));
    return v;
}

// ---------------------------------------------------------------------------
// Utility kernels
// ---------------------------------------------------------------------------
__global__ __launch_bounds__(256) void zero_kernel(uint32_t* p, size_t nwords) {
    size_t i = (size_t)blockIdx.x * blockDim.x + threadIdx.x;
    if (i < nwords) p[i] = 0u;
}

__global__ __launch_bounds__(256) void cvt_bf16_kernel(const float* src, bf16* dst, int n) {
    int i = blockIdx.x * blockDim.x + threadIdx.x;
    if (i < n) dst[i] = (bf16)src[i];
}

// ---------------------------------------------------------------------------
// LayerNorm of masked rows -> bf16. One wave per row.
// ---------------------------------------------------------------------------
__global__ __launch_bounds__(256) void ln_kernel(
    const float* __restrict__ imgs, const float* __restrict__ caps,
    const int* __restrict__ img_lens, const int* __restrict__ cap_lens,
    const float* __restrict__ g1, const float* __restrict__ b1,
    const float* __restrict__ g2, const float* __restrict__ b2,
    const float* __restrict__ g3, const float* __restrict__ b3,
    bf16* __restrict__ lnq, bf16* __restrict__ lnk, bf16* __restrict__ lnv) {
    const int NQ = BC * WW;   // 2560
    const int NK = BI * RR;   // 2304
    int wave = blockIdx.x * (blockDim.x >> 5) + (threadIdx.x >> 5);
    int lane = threadIdx.x & 31;
    if (wave >= NQ + 2 * NK) return;

    const float *x, *g, *b;
    bf16* out;
    int valid;
    if (wave < NQ) {
        int c = wave / WW, w = wave % WW;
        x = caps + (size_t)wave * DD;
        valid = (w < cap_lens[c]);
        g = g1; b = b1; out = lnq + (size_t)wave * DD;
    } else if (wave < NQ + NK) {
        int m = wave - NQ;
        int i = m / RR, r = m % RR;
        x = imgs + (size_t)m * DD;
        valid = (r < img_lens[i]);
        g = g2; b = b2; out = lnk + (size_t)m * DD;
    } else {
        int m = wave - NQ - NK;
        int i = m / RR, r = m % RR;
        x = imgs + (size_t)m * DD;
        valid = (r < img_lens[i]);
        g = g3; b = b3; out = lnv + (size_t)m * DD;
    }

    float v[16];
    float s = 0.f;
#pragma unroll
    for (int t = 0; t < 16; ++t) {
        int d = lane + t * 32;
        float xv = valid ? x[d] : 0.f;
        v[t] = xv; s += xv;
    }
    float mu = wave_sum(s) * (1.f / DD);
    float q = 0.f;
#pragma unroll
    for (int t = 0; t < 16; ++t) { float dl = v[t] - mu; q += dl * dl; }
    float var = wave_sum(q) * (1.f / DD);
    float rs = rsqrtf(var + 1e-5f);
#pragma unroll
    for (int t = 0; t < 16; ++t) {
        int d = lane + t * 32;
        out[d] = (bf16)((v[t] - mu) * rs * g[d] + b[d]);
    }
}

// ---------------------------------------------------------------------------
// q/k/v projections: Y = LN(X) @ W^T + bias, via WMMA.
// Each wave: one M-tile x four N-tiles, software-pipelined K loop.
// jobs: q: 160*8 = 1280 ; k: 144*8 = 1152 ; v: 144*8 = 1152 ; total 3584
// ---------------------------------------------------------------------------
__global__ __launch_bounds__(256) void qkv_gemm_kernel(
    const bf16* __restrict__ lnq, const bf16* __restrict__ lnk, const bf16* __restrict__ lnv,
    const bf16* __restrict__ Wqb, const bf16* __restrict__ Wkb, const bf16* __restrict__ Wvb,
    const float* __restrict__ bq, const float* __restrict__ bk, const float* __restrict__ bv,
    bf16* __restrict__ qb, float* __restrict__ qf,
    bf16* __restrict__ kb, bf16* __restrict__ vT) {
    int wave = blockIdx.x * (blockDim.x >> 5) + (threadIdx.x >> 5);
    int lane = threadIdx.x & 31;

    const bf16 *A, *B;
    const float* bias;
    int seg, job;
    if (wave < 1280)       { seg = 0; job = wave;        A = lnq; B = Wqb; bias = bq; }
    else if (wave < 2432)  { seg = 1; job = wave - 1280; A = lnk; B = Wkb; bias = bk; }
    else                   { seg = 2; job = wave - 2432; A = lnv; B = Wvb; bias = bv; }

    int mt = job >> 3;          // 8 N-groups of 64 columns
    int ng = job & 7;
    int mBase = mt * 16, nBase = ng * 64;

    v8f z = {};
    v8f acc[4] = {z, z, z, z};

    // software-pipelined K loop (prefetch kt+1 before issuing kt's WMMAs)
    v16bf af = load_a_frag(A, DD, mBase, 0, lane);
    v16bf bf0 = load_b_frag(B, DD, 0, nBase + 0,  lane);
    v16bf bf1 = load_b_frag(B, DD, 0, nBase + 16, lane);
    v16bf bf2 = load_b_frag(B, DD, 0, nBase + 32, lane);
    v16bf bf3 = load_b_frag(B, DD, 0, nBase + 48, lane);
#pragma unroll
    for (int kt = 0; kt < 16; ++kt) {
        v16bf an, bn0, bn1, bn2, bn3;
        if (kt != 15) {
            int kn = (kt + 1) * 32;
            an  = load_a_frag(A, DD, mBase, kn, lane);
            bn0 = load_b_frag(B, DD, kn, nBase + 0,  lane);
            bn1 = load_b_frag(B, DD, kn, nBase + 16, lane);
            bn2 = load_b_frag(B, DD, kn, nBase + 32, lane);
            bn3 = load_b_frag(B, DD, kn, nBase + 48, lane);
        }
        acc[0] = wmma_bf16(af, bf0, acc[0]);
        acc[1] = wmma_bf16(af, bf1, acc[1]);
        acc[2] = wmma_bf16(af, bf2, acc[2]);
        acc[3] = wmma_bf16(af, bf3, acc[3]);
        af = an; bf0 = bn0; bf1 = bn1; bf2 = bn2; bf3 = bn3;
    }

    int rbase = mBase + (lane >> 4) * 8;
#pragma unroll
    for (int r = 0; r < 8; ++r) {
        int m = rbase + r;
        if (seg == 0) {
            int c = m / WW, w = m % WW;
#pragma unroll
            for (int j = 0; j < 4; ++j) {
                int col = nBase + j * 16 + (lane & 15);
                float val = acc[j][r] + bias[col];
                qf[(size_t)m * DD + col] = val;
                qb[((size_t)c * WP + w) * DD + col] = (bf16)val;
            }
        } else if (seg == 1) {
            int i = m / RR, rr = m % RR;
#pragma unroll
            for (int j = 0; j < 4; ++j) {
                int col = nBase + j * 16 + (lane & 15);
                kb[((size_t)i * RP + rr) * DD + col] = (bf16)(acc[j][r] + bias[col]);
            }
        } else {
            int i = m / RR, rr = m % RR;
#pragma unroll
            for (int j = 0; j < 4; ++j) {
                int col = nBase + j * 16 + (lane & 15);
                vT[((size_t)i * DD + col) * RK + rr] = (bf16)(acc[j][r] + bias[col]);
            }
        }
    }
}

// ---------------------------------------------------------------------------
// Fused per-(caption,image)-pair kernel. All intermediates in LDS.
// GEMMs: each wave owns 4 N-tiles x 3 M-tiles (12 accumulators); every
// B fragment (Wo / v^T, L2-resident) is loaded exactly once per block, and
// the K loop is software-pipelined to overlap loads with the 12 WMMAs.
// ---------------------------------------------------------------------------
#define SM_S_OFF    0
#define SM_CTX_OFF  (WP * RP * 4)                       // 9216
#define SM_ATTN_OFF (SM_CTX_OFF + WP * DD * 4)          // 107520
#define SM_CTXN_OFF (SM_ATTN_OFF + WP * RK * 2)         // 113664
#define SM_TOTAL    (SM_CTXN_OFF + WP * DD * 2)         // 162816 bytes

__global__ __launch_bounds__(256) void pair_kernel(
    const bf16* __restrict__ qb, const float* __restrict__ qf,
    const bf16* __restrict__ kb, const bf16* __restrict__ vT,
    const bf16* __restrict__ Wob,
    const float* __restrict__ bo, const float* __restrict__ g4, const float* __restrict__ b4,
    const int* __restrict__ img_lens, const int* __restrict__ cap_lens,
    float* __restrict__ out) {
    extern __shared__ char smem[];
    float* sS    = (float*)(smem + SM_S_OFF);     // [WP][RP]
    float* sCtx  = (float*)(smem + SM_CTX_OFF);   // [WP][DD]  (reused for 'out')
    bf16*  sAttn = (bf16*)(smem + SM_ATTN_OFF);   // [WP][RK]
    bf16*  sCtxn = (bf16*)(smem + SM_CTXN_OFF);   // [WP][DD]

    int c = blockIdx.x & (BC - 1);
    int i = blockIdx.x >> 6;
    int wave = threadIdx.x >> 5;
    int lane = threadIdx.x & 31;

    const bf16* qc  = qb + (size_t)c * WP * DD;
    const bf16* ki  = kb + (size_t)i * RP * DD;
    const bf16* vTi = vT + (size_t)i * DD * RK;

    // warm L2 with the first slabs (global_prefetch_b8)
    __builtin_prefetch(Wob + (size_t)threadIdx.x * 128, 0, 1);
    __builtin_prefetch(vTi + (size_t)threadIdx.x * 128, 0, 1);

    // ---- 1. S = q k^T * (1/sqrt(D)) : waves 0..2, one N-tile, 3 M-tiles --
    const float scale = 0.044194173824159216f;   // 1/sqrt(512)
    if (wave < 3) {
        int nt = wave;
        v8f a0 = {}, a1 = {}, a2 = {};
        v16bf bfr = load_b_frag(ki, DD, 0, nt * 16, lane);
        v16bf f0  = load_a_frag(qc, DD, 0,  0, lane);
        v16bf f1  = load_a_frag(qc, DD, 16, 0, lane);
        v16bf f2  = load_a_frag(qc, DD, 32, 0, lane);
#pragma unroll
        for (int kt = 0; kt < 16; ++kt) {
            v16bf bn, g0, g1, g2;
            if (kt != 15) {
                int kn = (kt + 1) * 32;
                bn = load_b_frag(ki, DD, kn, nt * 16, lane);
                g0 = load_a_frag(qc, DD, 0,  kn, lane);
                g1 = load_a_frag(qc, DD, 16, kn, lane);
                g2 = load_a_frag(qc, DD, 32, kn, lane);
            }
            a0 = wmma_bf16(f0, bfr, a0);
            a1 = wmma_bf16(f1, bfr, a1);
            a2 = wmma_bf16(f2, bfr, a2);
            bfr = bn; f0 = g0; f1 = g1; f2 = g2;
        }
        int col = nt * 16 + (lane & 15);
        int rb  = (lane >> 4) * 8;
#pragma unroll
        for (int r = 0; r < 8; ++r) {
            sS[(rb + r) * RP + col]      = a0[r] * scale;
            sS[(16 + rb + r) * RP + col] = a1[r] * scale;
            sS[(32 + rb + r) * RP + col] = a2[r] * scale;
        }
    }
    __syncthreads();

    // ---- 2. masked softmax over regions ---------------------------------
    int Lr = img_lens[i];
    for (int w = wave; w < WP; w += 8) {
        int c0 = lane, c1 = lane + 32;
        float x0 = (c0 < Lr) ? sS[w * RP + c0] : -1e30f;
        float x1 = (c1 < Lr) ? sS[w * RP + c1] : -1e30f;
        float m = wave_max(fmaxf(x0, x1));
        float e0 = (c0 < Lr) ? __expf(x0 - m) : 0.f;
        float e1 = (c1 < Lr) ? __expf(x1 - m) : 0.f;
        float inv = 1.f / wave_sum(e0 + e1);
        sAttn[w * RK + c0] = (bf16)(e0 * inv);
        sAttn[w * RK + c1] = (bf16)(e1 * inv);
    }
    __syncthreads();

    // ---- 3. ctx = attn @ v : 4 N-tiles x 3 M-tiles per wave, K = 64 -----
    {
        int ntBase = wave * 4;
        v8f z = {};
        v8f acc[12] = {z, z, z, z, z, z, z, z, z, z, z, z};
#pragma unroll
        for (int kt = 0; kt < 2; ++kt) {
            v16bf f0 = load_a_frag(sAttn, RK, 0,  kt * 32, lane);
            v16bf f1 = load_a_frag(sAttn, RK, 16, kt * 32, lane);
            v16bf f2 = load_a_frag(sAttn, RK, 32, kt * 32, lane);
#pragma unroll
            for (int j = 0; j < 4; ++j) {
                v16bf b = load_b_frag(vTi, RK, kt * 32, (ntBase + j) * 16, lane);
                acc[0 * 4 + j] = wmma_bf16(f0, b, acc[0 * 4 + j]);
                acc[1 * 4 + j] = wmma_bf16(f1, b, acc[1 * 4 + j]);
                acc[2 * 4 + j] = wmma_bf16(f2, b, acc[2 * 4 + j]);
            }
        }
#pragma unroll
        for (int j = 0; j < 4; ++j) {
            int col = (ntBase + j) * 16 + (lane & 15);
            int rb  = (lane >> 4) * 8;
#pragma unroll
            for (int r = 0; r < 8; ++r) {
                sCtx[(rb + r) * DD + col]      = acc[0 * 4 + j][r];
                sCtx[(16 + rb + r) * DD + col] = acc[1 * 4 + j][r];
                sCtx[(32 + rb + r) * DD + col] = acc[2 * 4 + j][r];
            }
        }
    }
    __syncthreads();

    // ---- 4. LayerNorm(g4,b4) of ctx rows -> bf16 ------------------------
    for (int w = wave; w < WP; w += 8) {
        float v[16];
        float s = 0.f;
#pragma unroll
        for (int t = 0; t < 16; ++t) {
            int d = lane + t * 32;
            float xv = sCtx[w * DD + d];
            v[t] = xv; s += xv;
        }
        float mu = wave_sum(s) * (1.f / DD);
        float q = 0.f;
#pragma unroll
        for (int t = 0; t < 16; ++t) { float dl = v[t] - mu; q += dl * dl; }
        float rs = rsqrtf(wave_sum(q) * (1.f / DD) + 1e-5f);
#pragma unroll
        for (int t = 0; t < 16; ++t) {
            int d = lane + t * 32;
            sCtxn[w * DD + d] = (bf16)((v[t] - mu) * rs * g4[d] + b4[d]);
        }
    }
    __syncthreads();

    // ---- 5. out = ctxn @ Wo^T + bo : 4 N x 3 M per wave, K=512, piped ---
    {
        int ntBase = wave * 4;
        v8f z = {};
        v8f acc[12] = {z, z, z, z, z, z, z, z, z, z, z, z};

        v16bf f0 = load_a_frag(sCtxn, DD, 0,  0, lane);
        v16bf f1 = load_a_frag(sCtxn, DD, 16, 0, lane);
        v16bf f2 = load_a_frag(sCtxn, DD, 32, 0, lane);
        v16bf b0 = load_b_frag(Wob, DD, 0, (ntBase + 0) * 16, lane);
        v16bf b1 = load_b_frag(Wob, DD, 0, (ntBase + 1) * 16, lane);
        v16bf b2 = load_b_frag(Wob, DD, 0, (ntBase + 2) * 16, lane);
        v16bf b3 = load_b_frag(Wob, DD, 0, (ntBase + 3) * 16, lane);
#pragma unroll
        for (int kt = 0; kt < 16; ++kt) {
            v16bf g0, g1, g2, c0, c1, c2, c3;
            if (kt != 15) {
                int kn = (kt + 1) * 32;
                g0 = load_a_frag(sCtxn, DD, 0,  kn, lane);
                g1 = load_a_frag(sCtxn, DD, 16, kn, lane);
                g2 = load_a_frag(sCtxn, DD, 32, kn, lane);
                c0 = load_b_frag(Wob, DD, kn, (ntBase + 0) * 16, lane);
                c1 = load_b_frag(Wob, DD, kn, (ntBase + 1) * 16, lane);
                c2 = load_b_frag(Wob, DD, kn, (ntBase + 2) * 16, lane);
                c3 = load_b_frag(Wob, DD, kn, (ntBase + 3) * 16, lane);
            }
            acc[0]  = wmma_bf16(f0, b0, acc[0]);
            acc[1]  = wmma_bf16(f0, b1, acc[1]);
            acc[2]  = wmma_bf16(f0, b2, acc[2]);
            acc[3]  = wmma_bf16(f0, b3, acc[3]);
            acc[4]  = wmma_bf16(f1, b0, acc[4]);
            acc[5]  = wmma_bf16(f1, b1, acc[5]);
            acc[6]  = wmma_bf16(f1, b2, acc[6]);
            acc[7]  = wmma_bf16(f1, b3, acc[7]);
            acc[8]  = wmma_bf16(f2, b0, acc[8]);
            acc[9]  = wmma_bf16(f2, b1, acc[9]);
            acc[10] = wmma_bf16(f2, b2, acc[10]);
            acc[11] = wmma_bf16(f2, b3, acc[11]);
            f0 = g0; f1 = g1; f2 = g2;
            b0 = c0; b1 = c1; b2 = c2; b3 = c3;
        }
#pragma unroll
        for (int j = 0; j < 4; ++j) {
            int col = (ntBase + j) * 16 + (lane & 15);
            float bz = bo[col];
            int rb = (lane >> 4) * 8;
#pragma unroll
            for (int r = 0; r < 8; ++r) {
                sCtx[(rb + r) * DD + col]      = acc[0 * 4 + j][r] + bz;
                sCtx[(16 + rb + r) * DD + col] = acc[1 * 4 + j][r] + bz;
                sCtx[(32 + rb + r) * DD + col] = acc[2 * 4 + j][r] + bz;
            }
        }
    }
    __syncthreads();

    // ---- 6. s = (out . q) / (||out|| + eps), masked ---------------------
    int Lw = cap_lens[c];
    for (int w = wave; w < WW; w += 8) {
        const float* qrow = qf + ((size_t)c * WW + w) * DD;
        float num = 0.f, d2 = 0.f;
#pragma unroll
        for (int t = 0; t < 16; ++t) {
            int d = lane + t * 32;
            float o = sCtx[w * DD + d];
            num += o * qrow[d];
            d2  += o * o;
        }
        num = wave_sum(num);
        d2  = wave_sum(d2);
        if (lane == 0) {
            float s = num / (sqrtf(d2) + 1e-8f);
            if (w >= Lw) s = -1.0f;
            out[((size_t)i * BC + c) * WW + w] = s;
        }
    }
}

// ---------------------------------------------------------------------------
// Host-side launcher
// ---------------------------------------------------------------------------
extern "C" void kernel_launch(void* const* d_in, const int* in_sizes, int n_in,
                              void* d_out, int out_size, void* d_ws, size_t ws_size,
                              hipStream_t stream) {
    (void)in_sizes; (void)n_in; (void)out_size; (void)ws_size;
    const float* imgs     = (const float*)d_in[0];
    const float* caps     = (const float*)d_in[1];
    const int*   img_lens = (const int*)d_in[2];
    const int*   cap_lens = (const int*)d_in[3];
    const float* Wq = (const float*)d_in[4];  const float* bq = (const float*)d_in[5];
    const float* Wk = (const float*)d_in[6];  const float* bk = (const float*)d_in[7];
    const float* Wv = (const float*)d_in[8];  const float* bv = (const float*)d_in[9];
    const float* Wo = (const float*)d_in[10]; const float* bo = (const float*)d_in[11];
    const float* g1 = (const float*)d_in[12]; const float* b1 = (const float*)d_in[13];
    const float* g2 = (const float*)d_in[14]; const float* b2 = (const float*)d_in[15];
    const float* g3 = (const float*)d_in[16]; const float* b3 = (const float*)d_in[17];
    const float* g4 = (const float*)d_in[18]; const float* b4 = (const float*)d_in[19];
    float* outp = (float*)d_out;

    char* ws = (char*)d_ws;
    // byte offsets (all sizes multiples of 256)
    bf16* Wqb = (bf16*)(ws + 0);                     // 512*512*2 = 524288
    bf16* Wkb = (bf16*)(ws + 524288);
    bf16* Wvb = (bf16*)(ws + 1048576);
    bf16* Wob = (bf16*)(ws + 1572864);
    bf16* lnq = (bf16*)(ws + 2097152);               // 2560*512*2 = 2621440
    bf16* lnk = (bf16*)(ws + 4718592);               // 2304*512*2 = 2359296
    bf16* lnv = (bf16*)(ws + 7077888);               // 2359296
    bf16* qb  = (bf16*)(ws + 9437184);               // 64*48*512*2 = 3145728
    bf16* kb  = (bf16*)(ws + 12582912);              // 3145728
    bf16* vT  = (bf16*)(ws + 15728640);              // 64*512*64*2 = 4194304
    float* qf = (float*)(ws + 19922944);             // 64*40*512*4 = 5242880
    // total: 25165824 bytes

    // 0) zero padded qb/kb/vT region (contiguous: 3145728+3145728+4194304 B)
    {
        size_t nwords = (3145728ull + 3145728ull + 4194304ull) / 4ull;  // 2621440
        zero_kernel<<<(unsigned)((nwords + 255) / 256), 256, 0, stream>>>(
            (uint32_t*)qb, nwords);
    }
    // 1) weights -> bf16
    cvt_bf16_kernel<<<1024, 256, 0, stream>>>(Wq, Wqb, 512 * 512);
    cvt_bf16_kernel<<<1024, 256, 0, stream>>>(Wk, Wkb, 512 * 512);
    cvt_bf16_kernel<<<1024, 256, 0, stream>>>(Wv, Wvb, 512 * 512);
    cvt_bf16_kernel<<<1024, 256, 0, stream>>>(Wo, Wob, 512 * 512);
    // 2) masked LayerNorms (7168 rows -> 896 blocks of 8 waves)
    ln_kernel<<<896, 256, 0, stream>>>(imgs, caps, img_lens, cap_lens,
                                       g1, b1, g2, b2, g3, b3, lnq, lnk, lnv);
    // 3) q/k/v projections (3584 wave-jobs -> 448 blocks of 8 waves)
    qkv_gemm_kernel<<<448, 256, 0, stream>>>(lnq, lnk, lnv, Wqb, Wkb, Wvb,
                                             bq, bk, bv, qb, qf, kb, vT);
    // 4) fused attention + out-projection + cosine, one block per (c,i) pair
    pair_kernel<<<BI * BC, 256, SM_TOTAL, stream>>>(qb, qf, kb, vT, Wob,
                                                    bo, g4, b4, img_lens, cap_lens,
                                                    outp);
}
